// ProductKeyMemory_867583394235
// MI455X (gfx1250) — compile-verified, matrix operand order checked by reading, not run on hardware
//
#include <hip/hip_runtime.h>
#include <hip/hip_bf16.h>
#include <math.h>

typedef __bf16 bf16_t;
typedef __attribute__((ext_vector_type(16))) __bf16 v16bf;
typedef __attribute__((ext_vector_type(8)))  __bf16 v8bf;
typedef __attribute__((ext_vector_type(8)))  float  v8f;

#define N_SUBKEYS 256
#define D_KEY     128
#define D_MODEL   1024
#define TOPK      8

// ---------- bf16 helpers (round-to-nearest-even via bit math) ----------
__device__ __forceinline__ unsigned short bfbits(float f) {
  unsigned u = __float_as_uint(f);
  unsigned r = u + 0x7FFFu + ((u >> 16) & 1u);
  return (unsigned short)(r >> 16);
}
__device__ __forceinline__ bf16_t f2bf(float f) {
  unsigned short h = bfbits(f);
  bf16_t b; __builtin_memcpy(&b, &h, 2); return b;
}
__device__ __forceinline__ float bf2f(bf16_t b) {
  unsigned short h; __builtin_memcpy(&h, &b, 2);
  return __uint_as_float(((unsigned)h) << 16);
}

// ---------- order-preserving float<->uint key for packed argmax ----------
__device__ __forceinline__ unsigned fkey(float f) {
  unsigned u = __float_as_uint(f);
  return (u & 0x80000000u) ? ~u : (u | 0x80000000u);
}
__device__ __forceinline__ float funkey(unsigned k) {
  unsigned u = (k & 0x80000000u) ? (k & 0x7FFFFFFFu) : ~k;
  return __uint_as_float(u);
}

// ---------- WMMA bf16 fragment loader ----------
// A (16x32, MxK): lane<16 holds row M=lane, K in {kb..kb+7} U {kb+16..kb+23};
// lane>=16 holds row M=lane-16, K in {kb+8..kb+15} U {kb+24..kb+31}.
// B (32x16, KxN) mirrors this with N over lanes; since B[k][n] = W[n][k] we
// load W rows (row = n) with the same pattern.
__device__ __forceinline__ v16bf load_frag(const bf16_t* base, int stride,
                                           int row_base, int kbase, int lane) {
  int r    = row_base + (lane & 15);
  int koff = (lane >> 4) << 3;
  const bf16_t* p = base + r * stride + kbase + koff;
  v8bf lo = *(const v8bf*)(p);
  v8bf hi = *(const v8bf*)(p + 16);
  return __builtin_shufflevector(lo, hi, 0,1,2,3,4,5,6,7,8,9,10,11,12,13,14,15);
}

// ---------- wave-level top-8 of 256 scores (one wave per token) ----------
__device__ __forceinline__ void topk256(const float* sc_row, int lane,
                                        float* vout, int* iout) {
  float v[8];
#pragma unroll
  for (int i = 0; i < 8; ++i) v[i] = sc_row[lane + 32 * i];
#pragma unroll
  for (int r = 0; r < TOPK; ++r) {
    float bv = -3.0e38f; int bs = 0;
#pragma unroll
    for (int i = 0; i < 8; ++i) if (v[i] > bv) { bv = v[i]; bs = i; }
    unsigned long long p =
        ((unsigned long long)fkey(bv) << 32) | (unsigned)(bs * 32 + lane);
#pragma unroll
    for (int off = 16; off > 0; off >>= 1) {
      unsigned long long o = __shfl_xor(p, off, 32);
      if (o > p) p = o;
    }
    int widx = (int)(p & 0xFFFFFFFFu);          // c = slot*32 + lane
    if (lane == 0) { vout[r] = funkey((unsigned)(p >> 32)); iout[r] = widx; }
    if (lane == (widx & 31)) v[widx >> 5] = -3.0e38f;
  }
}

// =====================================================================
// Kernel 0: one-time f32 -> bf16 conversion of Wq, keys_a, keys_b
// =====================================================================
__global__ void pkm_convert(const float* __restrict__ Wq,
                            const float* __restrict__ ka,
                            const float* __restrict__ kb,
                            bf16_t* __restrict__ wqb,
                            bf16_t* __restrict__ kab,
                            bf16_t* __restrict__ kbb) {
  const int WQN = 2 * D_KEY * D_MODEL;      // 262144
  const int KN  = N_SUBKEYS * D_KEY;        // 32768
  int i = blockIdx.x * blockDim.x + threadIdx.x;
  int total = WQN + 2 * KN;
  for (; i < total; i += gridDim.x * blockDim.x) {
    if (i < WQN)            wqb[i]            = f2bf(Wq[i]);
    else if (i < WQN + KN)  kab[i - WQN]      = f2bf(ka[i - WQN]);
    else                    kbb[i - WQN - KN] = f2bf(kb[i - WQN - KN]);
  }
}

// =====================================================================
// Kernel 1: per 16-token tile: q = x*Wq^T (WMMA), scores (WMMA),
//           top-8 x2, combine top-8, softmax, gate.
// =====================================================================
__global__ __launch_bounds__(256) void pkm_scores_topk(
    const float* __restrict__ x, const bf16_t* __restrict__ wqb,
    const bf16_t* __restrict__ kab, const bf16_t* __restrict__ kbb,
    const float* __restrict__ gate_w, const float* __restrict__ gate_b,
    int* __restrict__ idx_out, float* __restrict__ w_out,
    float* __restrict__ g_out) {
  __shared__ __align__(16) bf16_t xs[16 * D_MODEL];     // 32 KB
  __shared__ __align__(16) bf16_t qs[16 * 2 * D_KEY];   //  8 KB
  __shared__ __align__(16) float  sc[16 * N_SUBKEYS];   // 16 KB (a then b)
  __shared__ float vA[16][TOPK]; __shared__ int iA[16][TOPK];
  __shared__ float vB[16][TOPK]; __shared__ int iB[16][TOPK];

  const int tid  = threadIdx.x;
  const int lane = tid & 31;
  const int wave = tid >> 5;
  const long t0  = (long)blockIdx.x * 16;

  // ---- Phase 0: x tile -> LDS as bf16 (tokens are contiguous rows) ----
  {
    const float4* xg = (const float4*)(x + t0 * D_MODEL);
    uint2* xs2 = (uint2*)xs;
    for (int e = tid; e < 16 * D_MODEL / 4; e += 256) {
      float4 v = xg[e];
      uint2 u;
      u.x = ((unsigned)bfbits(v.y) << 16) | bfbits(v.x);
      u.y = ((unsigned)bfbits(v.w) << 16) | bfbits(v.z);
      xs2[e] = u;
    }
  }
  __syncthreads();

  // ---- Phase 1: q[16x256] = x_tile[16x1024] @ Wq^T (bf16 WMMA) ----
  for (int nt = wave * 2; nt < wave * 2 + 2; ++nt) {
    v8f acc = {};
    for (int kb = 0; kb < D_MODEL; kb += 32) {
      v16bf a = load_frag(xs, D_MODEL, 0, kb, lane);
      v16bf b = load_frag(wqb, D_MODEL, nt * 16, kb, lane);
      acc = __builtin_amdgcn_wmma_f32_16x16x32_bf16(false, a, false, b,
                                                    (short)0, acc, false, false);
    }
    int n  = nt * 16 + (lane & 15);
    int mo = (lane >> 4) * 8;              // C/D layout: M = vgpr + 8*(lane>=16)
#pragma unroll
    for (int r = 0; r < 8; ++r) qs[(mo + r) * (2 * D_KEY) + n] = f2bf(acc[r]);
  }
  __syncthreads();

  // ---- Phase 2a: scores_a[16x256] = qa[16x128] @ keys_a^T ----
  for (int nt = wave * 2; nt < wave * 2 + 2; ++nt) {
    v8f acc = {};
    for (int kb = 0; kb < D_KEY; kb += 32) {
      v16bf a = load_frag(qs, 2 * D_KEY, 0, kb, lane);
      v16bf b = load_frag(kab, D_KEY, nt * 16, kb, lane);
      acc = __builtin_amdgcn_wmma_f32_16x16x32_bf16(false, a, false, b,
                                                    (short)0, acc, false, false);
    }
    int n  = nt * 16 + (lane & 15);
    int mo = (lane >> 4) * 8;
#pragma unroll
    for (int r = 0; r < 8; ++r) sc[(mo + r) * N_SUBKEYS + n] = acc[r];
  }
  __syncthreads();

  // ---- Phase 3a: top-8 of scores_a, one wave per token ----
  for (int tt = wave * 2; tt < wave * 2 + 2; ++tt)
    topk256(&sc[tt * N_SUBKEYS], lane, vA[tt], iA[tt]);
  __syncthreads();

  // ---- Phase 2b: scores_b (qb = q columns 128..255), reuse sc buffer ----
  for (int nt = wave * 2; nt < wave * 2 + 2; ++nt) {
    v8f acc = {};
    for (int kb = 0; kb < D_KEY; kb += 32) {
      v16bf a = load_frag(qs, 2 * D_KEY, 0, D_KEY + kb, lane);
      v16bf b = load_frag(kbb, D_KEY, nt * 16, kb, lane);
      acc = __builtin_amdgcn_wmma_f32_16x16x32_bf16(false, a, false, b,
                                                    (short)0, acc, false, false);
    }
    int n  = nt * 16 + (lane & 15);
    int mo = (lane >> 4) * 8;
#pragma unroll
    for (int r = 0; r < 8; ++r) sc[(mo + r) * N_SUBKEYS + n] = acc[r];
  }
  __syncthreads();

  // ---- Phase 3b: top-8 of scores_b ----
  for (int tt = wave * 2; tt < wave * 2 + 2; ++tt)
    topk256(&sc[tt * N_SUBKEYS], lane, vB[tt], iB[tt]);
  __syncthreads();

  // ---- Phase 4: 64-candidate combine, top-8, softmax -> weights/indices ----
  for (int tt = wave * 2; tt < wave * 2 + 2; ++tt) {
    float cv[2];
    int c0 = lane, c1 = lane + 32;
    cv[0] = vA[tt][c0 >> 3] + vB[tt][c0 & 7];
    cv[1] = vA[tt][c1 >> 3] + vB[tt][c1 & 7];
    float tv[8]; int tc[8];
#pragma unroll
    for (int r = 0; r < TOPK; ++r) {
      float bv; int bs;
      if (cv[0] >= cv[1]) { bv = cv[0]; bs = 0; } else { bv = cv[1]; bs = 1; }
      unsigned long long p =
          ((unsigned long long)fkey(bv) << 32) | (unsigned)(bs * 32 + lane);
#pragma unroll
      for (int off = 16; off > 0; off >>= 1) {
        unsigned long long o = __shfl_xor(p, off, 32);
        if (o > p) p = o;
      }
      int wc = (int)(p & 63);                 // candidate id 0..63
      tv[r] = funkey((unsigned)(p >> 32));
      tc[r] = wc;
      if (lane == ((int)(p & 0xFFFFFFFFu) & 31)) cv[(int)(p & 0xFFFFFFFFu) >> 5] = -3.0e38f;
    }
    if (lane == 0) {
      float m = tv[0], s = 0.f, e[8];
#pragma unroll
      for (int r = 0; r < TOPK; ++r) { e[r] = __expf(tv[r] - m); s += e[r]; }
      float inv = 1.0f / s;
#pragma unroll
      for (int r = 0; r < TOPK; ++r) {
        w_out[(t0 + tt) * TOPK + r]   = e[r] * inv;
        idx_out[(t0 + tt) * TOPK + r] =
            iA[tt][tc[r] >> 3] * N_SUBKEYS + iB[tt][tc[r] & 7];
      }
    }
  }

  // ---- Phase 5: gate = sigmoid(x . gate_w + b) (xs still intact) ----
  for (int tt = wave * 2; tt < wave * 2 + 2; ++tt) {
    float s = 0.f;
    for (int i = lane; i < D_MODEL; i += 32)
      s += bf2f(xs[tt * D_MODEL + i]) * gate_w[i];
#pragma unroll
    for (int off = 16; off > 0; off >>= 1) s += __shfl_xor(s, off, 32);
    if (lane == 0)
      g_out[t0 + tt] = 1.0f / (1.0f + __expf(-(s + gate_b[0])));
  }
}

// =====================================================================
// Kernel 2: out = x + g * sum_k w_k * values[idx_k]   (pure bandwidth)
// one block per token, 256 threads * float4 = 1024 floats
// =====================================================================
__global__ __launch_bounds__(256) void pkm_output(
    const float* __restrict__ x, const float* __restrict__ values,
    const int* __restrict__ idx, const float* __restrict__ w,
    const float* __restrict__ g, float* __restrict__ out) {
  const long t = blockIdx.x;
  const int  l = threadIdx.x;
  const float4* xv = (const float4*)(x + t * D_MODEL);
  float4 xd = xv[l];
  float4 acc = make_float4(0.f, 0.f, 0.f, 0.f);
#pragma unroll
  for (int k = 0; k < TOPK; ++k) {
    int   id = idx[t * TOPK + k];
    float wk = w[t * TOPK + k];
    const float4* vr = (const float4*)(values + (size_t)id * D_MODEL);
    float4 v = vr[l];
    acc.x += wk * v.x; acc.y += wk * v.y; acc.z += wk * v.z; acc.w += wk * v.w;
  }
  float gt = g[t];
  float4 o;
  o.x = xd.x + gt * acc.x; o.y = xd.y + gt * acc.y;
  o.z = xd.z + gt * acc.z; o.w = xd.w + gt * acc.w;
  ((float4*)(out + t * D_MODEL))[l] = o;
}

// =====================================================================
extern "C" void kernel_launch(void* const* d_in, const int* in_sizes, int n_in,
                              void* d_out, int out_size, void* d_ws, size_t ws_size,
                              hipStream_t stream) {
  const float* x      = (const float*)d_in[0];
  const float* ka     = (const float*)d_in[1];
  const float* kb     = (const float*)d_in[2];
  const float* values = (const float*)d_in[3];
  const float* Wq     = (const float*)d_in[4];
  const float* gw     = (const float*)d_in[5];
  const float* gbias  = (const float*)d_in[6];

  const int nTok = in_sizes[0] / D_MODEL;          // B*T = 8192

  // workspace carve-out
  bf16_t* wqb = (bf16_t*)d_ws;                      // 256*1024 bf16
  bf16_t* kab = wqb + 2 * D_KEY * D_MODEL;
  bf16_t* kbb = kab + N_SUBKEYS * D_KEY;
  int*    idxb = (int*)(kbb + N_SUBKEYS * D_KEY);   // nTok*8 ints
  float*  wtb  = (float*)(idxb + (size_t)nTok * TOPK);
  float*  gtb  = wtb + (size_t)nTok * TOPK;

  const int convTotal = 2 * D_KEY * D_MODEL + 2 * N_SUBKEYS * D_KEY;
  pkm_convert<<<(convTotal + 255) / 256, 256, 0, stream>>>(Wq, ka, kb, wqb, kab, kbb);
  pkm_scores_topk<<<nTok / 16, 256, 0, stream>>>(x, wqb, kab, kbb, gw, gbias,
                                                 idxb, wtb, gtb);
  pkm_output<<<nTok, 256, 0, stream>>>(x, values, idxb, wtb, gtb, (float*)d_out);
}
